// CustomAttentionModel_6150393168471
// MI455X (gfx1250) — matrix operand order repaired and checked by reading, not compiled, not run on hardware
//
#include <hip/hip_runtime.h>
#include <hip/hip_bf16.h>
#include <math.h>

#define D_MODEL 1024
#define NHEADS  16
#define HDIM    64
#define BATCH   2
#define SEQ     2048
#define MTOT    (BATCH * SEQ)   // 4096 tokens

typedef __attribute__((ext_vector_type(16))) __bf16 bf16x16;
typedef __attribute__((ext_vector_type(8)))  float  v8f;

// 16 bf16 = 32 bytes = two uint4 chunks
union Frag {
    bf16x16 bf;
    uint4   q[2];
};

__device__ __forceinline__ unsigned short f2bf(float x) {
    union { float f; unsigned u; } c; c.f = x;
    unsigned u = c.u;
    unsigned r = u + 0x7FFFu + ((u >> 16) & 1u);   // round-to-nearest-even
    return (unsigned short)(r >> 16);
}

// ---------------------------------------------------------------- converts
__global__ void k_f32_to_bf16(const float* __restrict__ in,
                              unsigned short* __restrict__ out, int n) {
    int i = blockIdx.x * blockDim.x + threadIdx.x;
    if (i < n) out[i] = f2bf(in[i]);
}

// W [K,N] row-major f32  ->  WT [N,K] row-major bf16
__global__ void k_transpose_bf16(const float* __restrict__ in,
                                 unsigned short* __restrict__ out) {
    int i = blockIdx.x * blockDim.x + threadIdx.x;
    if (i < D_MODEL * D_MODEL) {
        int n = i >> 10, k = i & 1023;
        out[i] = f2bf(in[k * D_MODEL + n]);
    }
}

// ---------------------------------------------------------------- GEMM
// C[M=4096, N=1024] = A[M,K=1024](bf16, row-major) x BT[N,K](bf16, row-major) + bias
// mode 0: store bf16 row-major      (Q, K projections)
// mode 1: store bf16 as V^T [B,H,Hd,S]   (V projection)
// mode 2: store f32 row-major       (output projection)
__global__ __launch_bounds__(256)
void k_gemm_bf16(const unsigned short* __restrict__ A,
                 const unsigned short* __restrict__ BT,
                 const float* __restrict__ bias,
                 void* __restrict__ Cout, int mode) {
    const int lane  = threadIdx.x & 31;
    const int wave  = threadIdx.x >> 5;
    const int hi    = (lane >> 4) & 1;
    const int l16   = lane & 15;
    const int tileM = blockIdx.y * 128 + (wave & 3) * 32;
    const int tileN = blockIdx.x * 64  + (wave >> 2) * 32;

    v8f acc[2][2] = {};

    for (int k0 = 0; k0 < D_MODEL; k0 += 32) {
        Frag a[2], b[2];
        const int kbA = k0 + (hi ? 16 : 0);
#pragma unroll
        for (int i = 0; i < 2; ++i) {
            const int row = tileM + i * 16 + l16;
            const uint4* p = (const uint4*)(A + (size_t)row * D_MODEL + kbA);
            a[i].q[0] = p[0];      // K = kbA .. kbA+7
            a[i].q[1] = p[1];      // K = kbA+8 .. kbA+15
        }
        const int kbB = k0 + (hi ? 16 : 0);
#pragma unroll
        for (int j = 0; j < 2; ++j) {
            const int col = tileN + j * 16 + l16;
            const uint4* p = (const uint4*)(BT + (size_t)col * D_MODEL + kbB);
            b[j].q[0] = p[0];      // K = kbB .. kbB+15 contiguous
            b[j].q[1] = p[1];
        }
#pragma unroll
        for (int i = 0; i < 2; ++i)
#pragma unroll
            for (int j = 0; j < 2; ++j)
                acc[i][j] = __builtin_amdgcn_wmma_f32_16x16x32_bf16(
                    false, a[i].bf, false, b[j].bf, (short)0, acc[i][j],
                    false, false);
    }

#pragma unroll
    for (int i = 0; i < 2; ++i) {
        const int mbase = tileM + i * 16 + hi * 8;   // C layout: M = r + 8*hi
#pragma unroll
        for (int j = 0; j < 2; ++j) {
            const int n  = tileN + j * 16 + l16;
            const float bn = bias[n];
#pragma unroll
            for (int r = 0; r < 8; ++r) {
                const int   m = mbase + r;
                const float v = acc[i][j][r] + bn;
                if (mode == 2) {
                    ((float*)Cout)[(size_t)m * D_MODEL + n] = v;
                } else if (mode == 1) {
                    const int bb = m >> 11, s = m & (SEQ - 1);
                    const int h  = n >> 6,  d = n & (HDIM - 1);
                    ((unsigned short*)Cout)
                        [((size_t)(bb * NHEADS + h) * HDIM + d) * SEQ + s] = f2bf(v);
                } else {
                    ((unsigned short*)Cout)[(size_t)m * D_MODEL + n] = f2bf(v);
                }
            }
        }
    }
}

// ---------------------------------------------------------------- attention
// Flash-style online softmax. Block = 4 waves, each wave owns 16 q rows.
// Q,K: bf16 [B*S, D]; VT: bf16 [B,H,Hd,S]; O: bf16 [B*S, D]
__global__ __launch_bounds__(128)
void k_attention(const unsigned short* __restrict__ Q,
                 const unsigned short* __restrict__ K,
                 const unsigned short* __restrict__ VT,
                 unsigned short* __restrict__ O) {
    __shared__ unsigned short lds[4 * 16 * 32];   // per-wave 16x32 P tile
    const int lane = threadIdx.x & 31;
    const int wave = threadIdx.x >> 5;
    const int hi   = (lane >> 4) & 1;
    const int l16  = lane & 15;
    const int b    = blockIdx.z;
    const int h    = blockIdx.y;
    const int q0   = blockIdx.x * 64 + wave * 16;
    unsigned short* plds = lds + wave * 512;

    // Q A-fragments for the two d-steps (contraction over HDIM=64)
    Frag qf[2];
    {
        const int row = b * SEQ + q0 + l16;
        const unsigned short* qr = Q + (size_t)row * D_MODEL + h * HDIM;
#pragma unroll
        for (int j = 0; j < 2; ++j) {
            const int kb = j * 32 + (hi ? 16 : 0);
            const uint4* p = (const uint4*)(qr + kb);
            qf[j].q[0] = p[0];
            qf[j].q[1] = p[1];
        }
    }

    v8f   acc[4] = {};
    float mrow[8], lrow[8];
#pragma unroll
    for (int r = 0; r < 8; ++r) { mrow[r] = -1e30f; lrow[r] = 0.f; }

    const unsigned short* kb_ptr = K  + (size_t)(b * SEQ) * D_MODEL + h * HDIM;
    const unsigned short* vbase  = VT + (size_t)((b * NHEADS + h) * HDIM) * SEQ;
    const float scale = 0.125f;   // 1/sqrt(64)

    for (int j0 = 0; j0 < SEQ; j0 += 32) {
        // ---- scores: 2 key-subtiles x 2 d-steps = 4 WMMAs
        v8f s[2] = {};
#pragma unroll
        for (int kt = 0; kt < 2; ++kt) {
#pragma unroll
            for (int ds = 0; ds < 2; ++ds) {
                Frag kf;
                const int key = j0 + kt * 16 + l16;
                const int db  = ds * 32 + (hi ? 16 : 0);
                const uint4* p = (const uint4*)(kb_ptr + (size_t)key * D_MODEL + db);
                kf.q[0] = p[0]; kf.q[1] = p[1];
                s[kt] = __builtin_amdgcn_wmma_f32_16x16x32_bf16(
                    false, qf[ds].bf, false, kf.bf, (short)0, s[kt], false, false);
            }
        }

        // ---- online softmax (rows M = r + 8*hi, cols spread over 16 lanes)
        float tm[8];
#pragma unroll
        for (int r = 0; r < 8; ++r) {
            s[0][r] *= scale; s[1][r] *= scale;
            tm[r] = fmaxf(s[0][r], s[1][r]);
        }
#pragma unroll
        for (int r = 0; r < 8; ++r)
#pragma unroll
            for (int mk = 1; mk < 16; mk <<= 1)
                tm[r] = fmaxf(tm[r], __shfl_xor(tm[r], mk, 32));

        float p0[8], p1[8], rs[8];
#pragma unroll
        for (int r = 0; r < 8; ++r) {
            const float mn = fmaxf(mrow[r], tm[r]);
            const float c  = __expf(mrow[r] - mn);
            mrow[r] = mn;
            lrow[r] *= c;
#pragma unroll
            for (int t = 0; t < 4; ++t) acc[t][r] *= c;
            p0[r] = __expf(s[0][r] - mn);
            p1[r] = __expf(s[1][r] - mn);
            rs[r] = p0[r] + p1[r];
        }
#pragma unroll
        for (int r = 0; r < 8; ++r) {
#pragma unroll
            for (int mk = 1; mk < 16; mk <<= 1)
                rs[r] += __shfl_xor(rs[r], mk, 32);
            lrow[r] += rs[r];
        }

        // ---- P (C-layout f32) -> LDS bf16 -> A-fragment layout
        // DS ops are in-order within a wave; LDS region is wave-private.
#pragma unroll
        for (int r = 0; r < 8; ++r) {
            const int row = r + hi * 8;
            plds[row * 32 + l16]      = f2bf(p0[r]);
            plds[row * 32 + 16 + l16] = f2bf(p1[r]);
        }
        Frag pf;
        {
            const int kb = hi ? 16 : 0;
            const uint4* lp = (const uint4*)(plds + l16 * 32 + kb);
            pf.q[0] = lp[0];
            pf.q[1] = lp[1];
        }

        // ---- P x V: 4 d-tiles of 16 cols, contiguous loads from V^T rows
#pragma unroll
        for (int t = 0; t < 4; ++t) {
            Frag vf;
            const int d   = t * 16 + l16;
            const int kb2 = j0 + (hi ? 16 : 0);
            const uint4* p = (const uint4*)(vbase + (size_t)d * SEQ + kb2);
            vf.q[0] = p[0]; vf.q[1] = p[1];
            acc[t] = __builtin_amdgcn_wmma_f32_16x16x32_bf16(
                false, pf.bf, false, vf.bf, (short)0, acc[t], false, false);
        }
    }

    // ---- normalize and store attended (bf16, merged-head row-major)
#pragma unroll
    for (int t = 0; t < 4; ++t) {
#pragma unroll
        for (int r = 0; r < 8; ++r) {
            const int m = b * SEQ + q0 + r + hi * 8;
            const int n = h * HDIM + t * 16 + l16;
            O[(size_t)m * D_MODEL + n] = f2bf(acc[t][r] / lrow[r]);
        }
    }
}

// ---------------------------------------------------------------- launch
extern "C" void kernel_launch(void* const* d_in, const int* in_sizes, int n_in,
                              void* d_out, int out_size, void* d_ws, size_t ws_size,
                              hipStream_t stream) {
    (void)in_sizes; (void)n_in; (void)out_size; (void)ws_size;
    const float* x  = (const float*)d_in[0];
    const float* Wq = (const float*)d_in[1];
    const float* bq = (const float*)d_in[2];
    const float* Wk = (const float*)d_in[3];
    const float* bk = (const float*)d_in[4];
    const float* Wv = (const float*)d_in[5];
    const float* bv = (const float*)d_in[6];
    const float* Wo = (const float*)d_in[7];
    const float* bo = (const float*)d_in[8];

    char* ws = (char*)d_ws;
    size_t off = 0;
    const size_t szX = (size_t)MTOT * D_MODEL * 2;      // 8 MB bf16 activations
    const size_t szW = (size_t)D_MODEL * D_MODEL * 2;   // 2 MB bf16 weights
    unsigned short* xb  = (unsigned short*)(ws + off); off += szX;
    unsigned short* wqT = (unsigned short*)(ws + off); off += szW;
    unsigned short* wkT = (unsigned short*)(ws + off); off += szW;
    unsigned short* wvT = (unsigned short*)(ws + off); off += szW;
    unsigned short* woT = (unsigned short*)(ws + off); off += szW;
    unsigned short* qb  = (unsigned short*)(ws + off); off += szX;
    unsigned short* kb  = (unsigned short*)(ws + off); off += szX;
    unsigned short* vt  = (unsigned short*)(ws + off); off += szX;
    unsigned short* att = (unsigned short*)(ws + off); off += szX;

    // 1) precision conversion (+ weight transpose for contiguous B-fragments)
    {
        const int n = MTOT * D_MODEL;
        k_f32_to_bf16<<<(n + 255) / 256, 256, 0, stream>>>(x, xb, n);
        const int nw = D_MODEL * D_MODEL;
        k_transpose_bf16<<<(nw + 255) / 256, 256, 0, stream>>>(Wq, wqT);
        k_transpose_bf16<<<(nw + 255) / 256, 256, 0, stream>>>(Wk, wkT);
        k_transpose_bf16<<<(nw + 255) / 256, 256, 0, stream>>>(Wv, wvT);
        k_transpose_bf16<<<(nw + 255) / 256, 256, 0, stream>>>(Wo, woT);
    }

    // 2) Q/K/V projections (V stored transposed per head)
    dim3 ggrid(D_MODEL / 64, MTOT / 128);
    k_gemm_bf16<<<ggrid, 256, 0, stream>>>(xb, wqT, bq, (void*)qb, 0);
    k_gemm_bf16<<<ggrid, 256, 0, stream>>>(xb, wkT, bk, (void*)kb, 0);
    k_gemm_bf16<<<ggrid, 256, 0, stream>>>(xb, wvT, bv, (void*)vt, 1);

    // 3) flash attention
    dim3 agrid(SEQ / 64, NHEADS, BATCH);
    k_attention<<<agrid, 128, 0, stream>>>(qb, kb, vt, att);

    // 4) output projection -> f32
    k_gemm_bf16<<<ggrid, 256, 0, stream>>>(att, woT, bo, d_out, 2);
}
